// GCNII_12687333392405
// MI455X (gfx1250) — matrix operand order repaired and checked by reading, not compiled, not run on hardware
//
#include <hip/hip_runtime.h>
#include <hip/hip_bf16.h>

typedef _Float16 half16 __attribute__((ext_vector_type(16)));
typedef float    float8 __attribute__((ext_vector_type(8)));

// ---------------- degree / normalization ----------------

__global__ void deg_init_k(float* __restrict__ deg, int Nn) {
    int i = blockIdx.x * blockDim.x + threadIdx.x;
    if (i < Nn) deg[i] = 1.0f;              // self-loop contributes 1
}

__global__ void deg_count_k(const int* __restrict__ dst, float* __restrict__ deg, int E) {
    int e = blockIdx.x * blockDim.x + threadIdx.x;
    if (e < E) atomicAdd(&deg[dst[e]], 1.0f);
}

__global__ void rsqrt_k(float* __restrict__ deg, int Nn) {
    int i = blockIdx.x * blockDim.x + threadIdx.x;
    if (i < Nn) deg[i] = rsqrtf(deg[i]);    // deg >= 1 always (self loops)
}

// ---------------- propagation: P = D^-1/2 A_hat D^-1/2 X ----------------

// self-loop term: P[i] = dinv[i]^2 * X[i]   (full overwrite -> initializes P)
__global__ void prop_self_k(const float* __restrict__ X, const float* __restrict__ dinv,
                            float* __restrict__ P, int Nn) {
    int idx = blockIdx.x * blockDim.x + threadIdx.x;
    int i = idx >> 4, ch = idx & 15;
    if (i < Nn) {
        float w = dinv[i] * dinv[i];
        float4 v = ((const float4*)X)[(size_t)i * 16 + ch];
        float4 r; r.x = w * v.x; r.y = w * v.y; r.z = w * v.z; r.w = w * v.w;
        ((float4*)P)[(size_t)i * 16 + ch] = r;
    }
}

// edge term: P[dst] += dinv[src]*dinv[dst] * X[src]  (16 threads / edge, float4 each)
__global__ void prop_edge_k(const int* __restrict__ src, const int* __restrict__ dst,
                            const float* __restrict__ X, const float* __restrict__ dinv,
                            float* __restrict__ P, int E) {
    int idx = blockIdx.x * blockDim.x + threadIdx.x;
    int e = idx >> 4, ch = idx & 15;
    if (e < E) {
        int s = src[e], d = dst[e];
        float w = dinv[s] * dinv[d];
        float4 v = ((const float4*)X)[(size_t)s * 16 + ch];
        float* pb = P + (size_t)d * 64 + ch * 4;
        atomicAdd(pb + 0, w * v.x);
        atomicAdd(pb + 1, w * v.y);
        atomicAdd(pb + 2, w * v.z);
        atomicAdd(pb + 3, w * v.w);
    }
}

// ---------------- W fragment pre-pack: f32 [64,64] -> f16 WMMA B-fragments ----------
// Layout: for col_tile t (0..3), k-step s (0..1), lane L (0..31):
//   Wf[((t*2+s)*32 + L)*16 + j] = (f16) W[(s*32 + (L>>4)*8 + j + (j>=8 ? 8 : 0))*64 + t*16 + (L&15)]
// so each wave's B operand is one contiguous 32-byte half16 load.
__global__ void pack_w_k(const float* __restrict__ W, _Float16* __restrict__ Wf) {
    int tid  = threadIdx.x;          // 256 threads = (t,s,lane)
    int lane = tid & 31;
    int s    = (tid >> 5) & 1;
    int t    = tid >> 6;
    int kbB  = (lane >> 4) * 8;
    int n    = t * 16 + (lane & 15);
    _Float16* o = Wf + (size_t)tid * 16;
#pragma unroll
    for (int j = 0; j < 16; ++j)
        o[j] = (_Float16)W[(s * 32 + kbB + j + (j >= 8 ? 8 : 0)) * 64 + n];
}

// ---------------- WMMA GEMM: out = act( blend(A, B2) @ W ) ----------------
// A, B2 : [Nn, 64] f32 (BLEND: input = 0.9*A + 0.1*B2)
// Wf    : pre-packed f16 B fragments (see pack_w_k)
// One wave -> one 16x16 tile; block = 256 threads = 8 waves = 32 rows x 64 cols.
template <bool BLEND, bool RELU>
__global__ void gemm64_wmma(const float* __restrict__ A, const float* __restrict__ B2,
                            const _Float16* __restrict__ Wf, float* __restrict__ out,
                            int Nn) {
    const int lane = threadIdx.x & 31;
    const int wave = threadIdx.x >> 5;
    const int m0 = blockIdx.x * 32 + (wave >> 2) * 16;   // row tile
    const int ct = wave & 3;                             // col tile 0..3
    const int n0 = ct * 16;

    int r = m0 + (lane & 15);
    int rc = (r < Nn) ? r : (Nn - 1);                    // clamp loads, keep EXEC full
    const int kbA = (lane >> 4) * 8;                     // A: K sub-block per half-wave
    const int n   = n0 + (lane & 15);

    const float4* arow = (const float4*)(A  + (size_t)rc * 64);
    const float4* brow = (const float4*)(B2 + (size_t)rc * 64);

    float8 c = {};
#pragma unroll
    for (int s = 0; s < 2; ++s) {                        // K = 64 -> 2 steps of 32
        // A fragment: elems 0..7 = K = s*32+kbA+0..7, elems 8..15 = +16
        float4 a0 = arow[(s * 32 + kbA) >> 2];
        float4 a1 = arow[(s * 32 + kbA + 4) >> 2];
        float4 a2 = arow[(s * 32 + kbA + 16) >> 2];
        float4 a3 = arow[(s * 32 + kbA + 20) >> 2];
        if (BLEND) {
            float4 p0 = brow[(s * 32 + kbA) >> 2];
            float4 p1 = brow[(s * 32 + kbA + 4) >> 2];
            float4 p2 = brow[(s * 32 + kbA + 16) >> 2];
            float4 p3 = brow[(s * 32 + kbA + 20) >> 2];
            a0.x = fmaf(0.9f, a0.x, 0.1f * p0.x); a0.y = fmaf(0.9f, a0.y, 0.1f * p0.y);
            a0.z = fmaf(0.9f, a0.z, 0.1f * p0.z); a0.w = fmaf(0.9f, a0.w, 0.1f * p0.w);
            a1.x = fmaf(0.9f, a1.x, 0.1f * p1.x); a1.y = fmaf(0.9f, a1.y, 0.1f * p1.y);
            a1.z = fmaf(0.9f, a1.z, 0.1f * p1.z); a1.w = fmaf(0.9f, a1.w, 0.1f * p1.w);
            a2.x = fmaf(0.9f, a2.x, 0.1f * p2.x); a2.y = fmaf(0.9f, a2.y, 0.1f * p2.y);
            a2.z = fmaf(0.9f, a2.z, 0.1f * p2.z); a2.w = fmaf(0.9f, a2.w, 0.1f * p2.w);
            a3.x = fmaf(0.9f, a3.x, 0.1f * p3.x); a3.y = fmaf(0.9f, a3.y, 0.1f * p3.y);
            a3.z = fmaf(0.9f, a3.z, 0.1f * p3.z); a3.w = fmaf(0.9f, a3.w, 0.1f * p3.w);
        }
        half16 a;
        a[0]  = (_Float16)a0.x; a[1]  = (_Float16)a0.y;
        a[2]  = (_Float16)a0.z; a[3]  = (_Float16)a0.w;
        a[4]  = (_Float16)a1.x; a[5]  = (_Float16)a1.y;
        a[6]  = (_Float16)a1.z; a[7]  = (_Float16)a1.w;
        a[8]  = (_Float16)a2.x; a[9]  = (_Float16)a2.y;
        a[10] = (_Float16)a2.z; a[11] = (_Float16)a2.w;
        a[12] = (_Float16)a3.x; a[13] = (_Float16)a3.y;
        a[14] = (_Float16)a3.z; a[15] = (_Float16)a3.w;

        // B fragment: single contiguous 32B vector load from packed weights
        half16 b = *(const half16*)(Wf + (size_t)(((ct * 2 + s) * 32 + lane)) * 16);

        c = __builtin_amdgcn_wmma_f32_16x16x32_f16(false, a, false, b, (short)0, c,
                                                   false, false);
    }

    // C/D f32 layout: VGPR v -> M = m0 + (lane/16)*8 + v, N = n0 + lane%16
    const int mbase = m0 + (lane >> 4) * 8;
    float* obase = out + (size_t)mbase * 64 + n;
    if (m0 + 16 <= Nn) {
        // full tile (wave-uniform branch): 8 plain stores, constant offsets
#pragma unroll
        for (int v = 0; v < 8; ++v) {
            float val = c[v];
            if (RELU) val = fmaxf(val, 0.0f);
            obase[v * 64] = val;
        }
    } else {
#pragma unroll
        for (int v = 0; v < 8; ++v) {
            if (mbase + v < Nn) {
                float val = c[v];
                if (RELU) val = fmaxf(val, 0.0f);
                obase[v * 64] = val;
            }
        }
    }
}

// ---------------- heads: h = P + b_gcn; out1 = h@W1+b1; out2 = h@W2+b2 ----------------

__global__ void heads_k(const float* __restrict__ P, const float* __restrict__ bg,
                        const float* __restrict__ W1, const float* __restrict__ b1,
                        const float* __restrict__ W2, const float* __restrict__ b2,
                        float* __restrict__ out1, float* __restrict__ out2,
                        float* __restrict__ hout, int Nn) {
    int i = blockIdx.x * blockDim.x + threadIdx.x;
    if (i >= Nn) return;
    const float* p = P + (size_t)i * 64;
    float* ho = hout + (size_t)i * 64;
    float a0 = b1[0], a1 = b1[1], a2 = b1[2], a3 = b1[3];
    float c0 = b2[0], c1 = b2[1], c2 = b2[2];
#pragma unroll
    for (int f = 0; f < 64; ++f) {
        float h = p[f] + bg[f];
        ho[f] = h;
        a0 = fmaf(h, W1[f * 4 + 0], a0);
        a1 = fmaf(h, W1[f * 4 + 1], a1);
        a2 = fmaf(h, W1[f * 4 + 2], a2);
        a3 = fmaf(h, W1[f * 4 + 3], a3);
        c0 = fmaf(h, W2[f * 3 + 0], c0);
        c1 = fmaf(h, W2[f * 3 + 1], c1);
        c2 = fmaf(h, W2[f * 3 + 2], c2);
    }
    out1[(size_t)i * 4 + 0] = a0;
    out1[(size_t)i * 4 + 1] = a1;
    out1[(size_t)i * 4 + 2] = a2;
    out1[(size_t)i * 4 + 3] = a3;
    out2[(size_t)i * 3 + 0] = c0;
    out2[(size_t)i * 3 + 1] = c1;
    out2[(size_t)i * 3 + 2] = c2;
}

// ---------------- driver ----------------

extern "C" void kernel_launch(void* const* d_in, const int* in_sizes, int n_in,
                              void* d_out, int out_size, void* d_ws, size_t ws_size,
                              hipStream_t stream) {
    const float* x   = (const float*)d_in[0];
    const int*   ei  = (const int*)d_in[1];
    const float* Ws  = (const float*)d_in[2];
    const float* Wg  = (const float*)d_in[3];
    const float* bg  = (const float*)d_in[4];
    const float* W1  = (const float*)d_in[5];
    const float* b1  = (const float*)d_in[6];
    const float* W2  = (const float*)d_in[7];
    const float* b2  = (const float*)d_in[8];

    const int Nn  = in_sizes[0] / 64;           // 100000
    const int E   = in_sizes[1] / 2;            // 1000000
    const int Lm1 = in_sizes[2] / (64 * 64);    // 3

    float* ws   = (float*)d_ws;
    float* dinv = ws;                           // [Nn]
    float* P    = ws + Nn;                      // [Nn,64]
    float* Xa   = P + (size_t)Nn * 64;          // [Nn,64]
    float* Xb   = Xa + (size_t)Nn * 64;         // [Nn,64]
    // packed f16 weight fragments (8 KB), 64B-aligned
    uintptr_t wf_addr = ((uintptr_t)(Xb + (size_t)Nn * 64) + 63) & ~(uintptr_t)63;
    _Float16* Wf = (_Float16*)wf_addr;

    const int* src = ei;                        // edge_index[0]
    const int* dst = ei + E;                    // edge_index[1]

    dim3 blk(256);
    int gN   = (Nn + 255) / 256;
    int gE   = (E + 255) / 256;
    int gN16 = (Nn * 16 + 255) / 256;
    int gE16 = (E * 16 + 255) / 256;
    int gG   = (Nn + 31) / 32;

    // gcn_norm
    deg_init_k<<<gN, blk, 0, stream>>>(dinv, Nn);
    deg_count_k<<<gE, blk, 0, stream>>>(dst, dinv, E);
    rsqrt_k<<<gN, blk, 0, stream>>>(dinv, Nn);

    // GCN2Conv layers: x = relu((0.9*prop(x) + 0.1*x) @ Ws[l])
    const float* xcur = x;
    float* bufs[2] = {Xa, Xb};
    int pp = 0;
    for (int l = 0; l < Lm1; ++l) {
        prop_self_k<<<gN16, blk, 0, stream>>>(xcur, dinv, P, Nn);
        prop_edge_k<<<gE16, blk, 0, stream>>>(src, dst, xcur, dinv, P, E);
        pack_w_k<<<1, blk, 0, stream>>>(Ws + (size_t)l * 64 * 64, Wf);
        float* xnext = bufs[pp];
        gemm64_wmma<true, true><<<gG, blk, 0, stream>>>(P, xcur, Wf, xnext, Nn);
        xcur = xnext;
        pp ^= 1;
    }

    // GCNConv: h = prop(x @ W_gcn) + b_gcn (bias fused into heads)
    float* xw = bufs[pp];
    pack_w_k<<<1, blk, 0, stream>>>(Wg, Wf);
    gemm64_wmma<false, false><<<gG, blk, 0, stream>>>(xcur, xcur, Wf, xw, Nn);
    prop_self_k<<<gN16, blk, 0, stream>>>(xw, dinv, P, Nn);
    prop_edge_k<<<gE16, blk, 0, stream>>>(src, dst, xw, dinv, P, E);

    float* out1 = (float*)d_out;               // [Nn,4]
    float* out2 = out1 + (size_t)Nn * 4;       // [Nn,3]
    float* hout = out1 + (size_t)Nn * 7;       // [Nn,64]
    heads_k<<<gN, blk, 0, stream>>>(P, bg, W1, b1, W2, b2, out1, out2, hout, Nn);
}
